// LSTM_34686155882815
// MI455X (gfx1250) — matrix operand order repaired and checked by reading, not compiled, not run on hardware
//
#include <hip/hip_runtime.h>
#include <hip/hip_bf16.h>

// ---------------------------------------------------------------------------
// Problem dims (from reference): S=2048, B=64, I=H=O=256
// ---------------------------------------------------------------------------
#define S_LEN 2048
#define BATCH 64
#define IDIM  256
#define HDIM  256
#define ODIM  256

typedef __attribute__((ext_vector_type(16))) __bf16 v16bf;
typedef __attribute__((ext_vector_type(8)))  float  v8f;
typedef unsigned int u32x4 __attribute__((ext_vector_type(4)));
typedef int          i32x8 __attribute__((ext_vector_type(8)));
typedef int          i32x4 __attribute__((ext_vector_type(4)));

// ---------------------------------------------------------------------------
// Workspace layout (bytes). Total ~137 MB.
//   xb   : x converted to bf16                [S*B*I]
//   hs   : layer-1 hidden outputs (bf16)      [S*B*H]
//   wrec : recurrent weights, fragment-swizzled bf16, 16 WG slices x 65536 el
//   wlin : w_lin fragment-swizzled bf16, 65536 el
//   bias : fused biases f32: b0[1024] b1[1024] blin[256]
//   hbuf : h state double buffers bf16 [layer][parity][64][256]
// ---------------------------------------------------------------------------
#define WS_XB    ((size_t)0)
#define WS_HS    (WS_XB   + (size_t)S_LEN*BATCH*IDIM*2)
#define WS_WREC  (WS_HS   + (size_t)S_LEN*BATCH*HDIM*2)
#define WS_WLIN  (WS_WREC + (size_t)16*65536*2)
#define WS_BIAS  (WS_WLIN + (size_t)65536*2)
#define WS_HBUF  (WS_BIAS + (size_t)(1024+1024+256)*4)

// ---------------------------------------------------------------------------
// WMMA fragment helpers (contiguous k = 16*(lane/16) + e for A and B, wave32)
// ---------------------------------------------------------------------------
__device__ __forceinline__ v16bf frag_join(u32x4 lo, u32x4 hi) {
    union { u32x4 q[2]; v16bf v; } u;
    u.q[0] = lo; u.q[1] = hi;
    return u.v;
}

// A fragment (16x32 bf16, M x K): lane l (m=l&15, kh=l>>4) holds
// K = kh*16..kh*16+15 of row m.  A is staged
// row-major in LDS with 264-element rows (528B: 16B-aligned, 4-bank skew
// per row -> conflict-free ds_load_b128).
__device__ __forceinline__ v16bf load_afrag(const __bf16* A264, int mt, int ktl, int lane) {
    const __bf16* p = A264 + (mt*16 + (lane & 15))*264 + ktl*32 + (lane >> 4)*16;
    return frag_join(*(const u32x4*)p, *(const u32x4*)(p + 8));
}

// B fragment (32x16 bf16, K x N): lane l (n=l&15, kh=l>>4) holds
// K = kh*16..kh*16+15 of column n.  Weights are pre-swizzled offline so each
// fragment is 1024 contiguous bytes; lane reads its 32B at lane*32.
__device__ __forceinline__ v16bf load_bfrag(const __bf16* Wl, int frag, int lane) {
    const u32x4* p = (const u32x4*)(Wl + frag*512);
    return frag_join(p[lane*2], p[lane*2 + 1]);
}

__device__ __forceinline__ float sigmoidf_(float x) { return 1.f / (1.f + __expf(-x)); }

// ---------------------------------------------------------------------------
// TDM: async tensor load of a 64x256 bf16 tile (row-major, contiguous in
// global) into LDS with 16B padding after every 512B row -> 264-element rows.
// Descriptor packing per CDNA5 ISA.
// This toolchain exposes the 6-arg builtin:
//   (u32x4 g0, i32x8 g1, i32x4 g2, i32x4 g3, i32x8 g4, i32 cpol)
// Groups 2/3/4 are unused for a 2D tensor -> zero-filled.
// ---------------------------------------------------------------------------
#if __has_builtin(__builtin_amdgcn_tensor_load_to_lds)
#define HAVE_TDM 1
#else
#define HAVE_TDM 0
#endif

__device__ __forceinline__ void tdm_stage_64x256(const __bf16* gsrc, const void* ldsdst) {
#if HAVE_TDM
    unsigned long long ga = (unsigned long long)(const void*)gsrc;
    unsigned int ldsoff = (unsigned int)((unsigned long long)ldsdst & 0xffffffffull);
    u32x4 g0;
    g0[0] = 1u;                                                   // count=1 (valid D#)
    g0[1] = ldsoff;                                               // lds_addr
    g0[2] = (unsigned int)(ga & 0xffffffffull);                   // global_addr[31:0]
    g0[3] = (unsigned int)((ga >> 32) & 0x01ffffffu) | (2u << 30);// global_addr[56:32] | type=2
    i32x8 g1;
    // data_size=1 (2B), pad_enable, pad_interval=6 (128 DW = 512B),
    // pad_amount=3 (4 DW = 16B); workgroup_mask=0 (no multicast)
    g1[0] = (int)((1u << 16) | (1u << 20) | (6u << 22) | (3u << 25));
    g1[1] = (int)(256u << 16);          // tensor_dim0[15:0]=256
    g1[2] = (int)(0u | (64u << 16));    // tensor_dim0 hi | tensor_dim1[15:0]=64
    g1[3] = (int)(0u | (256u << 16));   // tensor_dim1 hi | tile_dim0=256
    g1[4] = (int)(64u);                 // tile_dim1=64, tile_dim2=0
    g1[5] = 256;                        // tensor_dim0_stride[31:0]=256 elements
    g1[6] = 0;                          // stride0 hi | stride1 lo
    g1[7] = 0;                          // stride1 hi
    i32x4 z4 = {0, 0, 0, 0};            // 2D tensor: groups 2/3 unused
    i32x8 z8 = {0, 0, 0, 0, 0, 0, 0, 0};
    __builtin_amdgcn_tensor_load_to_lds(g0, g1, z4, z4, z8, 0);
#endif
}

// One TDM issue per workgroup (wave 0; TDM ignores EXEC, one op per wave).
__device__ __forceinline__ void stage_tile(__bf16* dst264, const __bf16* src, int tid, int wv) {
#if HAVE_TDM
    if (wv == 0) tdm_stage_64x256(src, dst264);
#else
    for (int i = tid; i < BATCH*32; i += 256) {
        int r = i >> 5, c = i & 31;
        ((u32x4*)(dst264 + r*264))[c] = ((const u32x4*)(src + r*256))[c];
    }
#endif
}
__device__ __forceinline__ void stage_wait(int wv) {
#if HAVE_TDM
    if (wv == 0) __builtin_amdgcn_s_wait_tensorcnt(0);
#endif
    __syncthreads();
}

// Cluster barrier with store-release / load-acquire for cross-WGP h exchange.
__device__ __forceinline__ void cluster_release() {
    asm volatile("s_wait_storecnt 0x0" ::: "memory");
}
__device__ __forceinline__ void cluster_barrier_acq() {
    __builtin_amdgcn_s_cluster_barrier();
    // WGP$ is read-only: stores already hit device L2; invalidate local lines
    // so next h loads observe other WGPs' writes.
    asm volatile("global_inv scope:SCOPE_DEV" ::: "memory");
    asm volatile("s_wait_loadcnt 0x0" ::: "memory");
}

// ---------------------------------------------------------------------------
// Prep kernels
// ---------------------------------------------------------------------------
__global__ void cvt_x_kernel(const float* __restrict__ x, __bf16* __restrict__ xb, int n) {
    for (int i = blockIdx.x*blockDim.x + threadIdx.x; i < n; i += gridDim.x*blockDim.x)
        xb[i] = (__bf16)x[i];
}

// Pre-swizzle recurrent weights into per-WG B-fragment order.
// Layer L, WG wgl owns h-cols [wgl*32, wgl*32+32); gate cols q*256 + those.
// Combined K=512: k<256 -> w_ih[gc][k], else w_hh[gc][k-256].
// Fragment f = (q*2+nt)*16+kt, 512 elements: lane(0..31) x e(0..15):
//   gc = q*256 + wgl*32 + nt*16 + (lane&15);  k = kt*32 + (lane>>4)*16 + e
__global__ void pack_rec_kernel(const float* __restrict__ w_ih0, const float* __restrict__ w_hh0,
                                const float* __restrict__ w_ih1, const float* __restrict__ w_hh1,
                                __bf16* __restrict__ wrec) {
    int idx = blockIdx.x*256 + threadIdx.x;
    if (idx >= 2*524288) return;
    int L   = idx >> 19;
    int r   = idx & 524287;
    int wgl = r >> 16;
    int r2  = r & 65535;
    int f   = r2 >> 9;
    int e9  = r2 & 511;
    int lane = e9 >> 4, e = e9 & 15;
    int q  = f >> 5;
    int nt = (f >> 4) & 1;
    int kt = f & 15;
    int gc = q*256 + wgl*32 + nt*16 + (lane & 15);
    int k  = kt*32 + (lane >> 4)*16 + e;
    const float* wih = L ? w_ih1 : w_ih0;
    const float* whh = L ? w_hh1 : w_hh0;
    float v = (k < 256) ? wih[gc*256 + k] : whh[gc*256 + (k - 256)];
    wrec[idx] = (__bf16)v;
}

// w_lin fragments (f = ntile*8 + kt) + fused biases.
__global__ void pack_misc_kernel(const float* __restrict__ w_lin,
                                 const float* __restrict__ b_ih0, const float* __restrict__ b_hh0,
                                 const float* __restrict__ b_ih1, const float* __restrict__ b_hh1,
                                 const float* __restrict__ b_lin,
                                 __bf16* __restrict__ wlin, float* __restrict__ bias) {
    int idx = blockIdx.x*256 + threadIdx.x;
    if (idx < 65536) {
        int f = idx >> 9, e9 = idx & 511;
        int lane = e9 >> 4, e = e9 & 15;
        int ntl = f >> 3, kt = f & 7;
        int gc = ntl*16 + (lane & 15);
        int k  = kt*32 + (lane >> 4)*16 + e;
        wlin[idx] = (__bf16)w_lin[gc*256 + k];
    }
    if (idx < 1024)      bias[idx] = b_ih0[idx] + b_hh0[idx];
    else if (idx < 2048) bias[idx] = b_ih1[idx - 1024] + b_hh1[idx - 1024];
    else if (idx < 2304) bias[idx] = b_lin[idx - 2048];
}

// ---------------------------------------------------------------------------
// Persistent recurrent kernel: one 16-WG cluster (8 WGs layer0, 8 WGs layer1).
// Each WG: 128 KB weight slice in LDS, 8 waves, wave wv -> gate q=wv>>1,
// n-tile nt=wv&1, all 4 M-tiles.  512 WMMAs per WG per step (K=512).
// NOTE: on hardware this must be dispatched as one 16-workgroup cluster so
// s_barrier_signal/wait -3 synchronize across WGPs.
// LDS: Wl 131072 + Ax 33792 + Ah 33792 + Gb 32768 + Cb 8192 = 239616 B
// ---------------------------------------------------------------------------
__global__ __launch_bounds__(256, 1)
void lstm_recurrent_kernel(const __bf16* __restrict__ xb,
                           __bf16* __restrict__ hs,
                           const __bf16* __restrict__ wrec,
                           const float* __restrict__ bias,
                           __bf16* __restrict__ hbuf) {
    extern __shared__ char smem[];
    __bf16* Wl = (__bf16*)smem;                                   // 131072 B
    __bf16* Ax = (__bf16*)(smem + 131072);                        // 33792 B
    __bf16* Ah = (__bf16*)(smem + 131072 + 33792);                // 33792 B
    float*  Gb = (float*)(smem + 131072 + 2*33792);               // 32768 B
    float*  Cb = (float*)(smem + 131072 + 2*33792 + 32768);       // 8192 B

    const int tid  = threadIdx.x;
    const int lane = tid & 31;
    const int wv   = tid >> 5;
    const int wg   = blockIdx.x;            // 0..15
    const bool L1  = wg >= 8;
    const int wgl  = L1 ? (wg - 8) : wg;
    const int q    = wv >> 1;
    const int nt   = wv & 1;

    // Load this WG's swizzled weight slice into LDS.
    {
        const u32x4* src = (const u32x4*)(wrec + (size_t)wg*65536);
        u32x4* dst = (u32x4*)Wl;
        for (int i = tid; i < 8192; i += 256) dst[i] = src[i];
    }
    // Zero cell state + this WG's h slice in both parity buffers.
    const size_t lbase = (size_t)(L1 ? 1 : 0) * 2 * BATCH * HDIM;
    for (int i = tid; i < BATCH*32; i += 256) Cb[i] = 0.f;
    for (int p = 0; p < 2; ++p)
        for (int i = tid; i < BATCH*32; i += 256) {
            int r = i >> 5, c = i & 31;
            hbuf[lbase + (size_t)p*BATCH*HDIM + r*HDIM + wgl*32 + c] = (__bf16)0.f;
        }
    __syncthreads();
    cluster_release();
    cluster_barrier_acq();

    const float* bl = bias + (L1 ? 1024 : 0);

    for (int t = 0; t < S_LEN; ++t) {
        const int par = t & 1, prv = par ^ 1;
        __bf16* hcur0            = hbuf + (size_t)par*BATCH*HDIM;            // layer0 h[t]
        const __bf16* hprv_own   = hbuf + lbase + (size_t)prv*BATCH*HDIM;    // own layer h[t-1]
        __bf16* hcur_own         = hbuf + lbase + (size_t)par*BATCH*HDIM;    // own layer h[t]

        v8f acc[4] = {};

        if (!L1) {
            // ---- layer 0: gates = [x_t | h0[t-1]] @ W0^T ----
            stage_tile(Ax, xb + (size_t)t*BATCH*IDIM, tid, wv);
            stage_tile(Ah, hprv_own, tid, wv);
            stage_wait(wv);
#pragma unroll 4
            for (int kt = 0; kt < 16; ++kt) {
                const __bf16* A = (kt < 8) ? Ax : Ah;
                v16bf b = load_bfrag(Wl, (q*2 + nt)*16 + kt, lane);
#pragma unroll
                for (int m = 0; m < 4; ++m) {
                    v16bf a = load_afrag(A, m, kt & 7, lane);
                    acc[m] = __builtin_amdgcn_wmma_f32_16x16x32_bf16(
                        false, a, false, b, (short)0, acc[m], false, false);
                }
            }
            { // C tiles -> LDS gate buffer
                int n = lane & 15, mh = lane >> 4;
#pragma unroll
                for (int m = 0; m < 4; ++m)
#pragma unroll
                    for (int r = 0; r < 8; ++r)
                        Gb[q*2048 + (m*16 + mh*8 + r)*32 + nt*16 + n] = acc[m][r];
            }
            __syncthreads();
            for (int i = tid; i < BATCH*32; i += 256) {
                int row = i >> 5, c = i & 31;
                int gcol = wgl*32 + c;
                float gi = Gb[0*2048 + i] + bl[0*256 + gcol];
                float gf = Gb[1*2048 + i] + bl[1*256 + gcol];
                float gg = Gb[2*2048 + i] + bl[2*256 + gcol];
                float go = Gb[3*2048 + i] + bl[3*256 + gcol];
                float cc = sigmoidf_(gf)*Cb[i] + sigmoidf_(gi)*tanhf(gg);
                Cb[i] = cc;
                hcur_own[row*HDIM + gcol] = (__bf16)(sigmoidf_(go)*tanhf(cc));
            }
            cluster_release();
            __syncthreads();
            cluster_barrier_acq();          // publish h0[t]
            __syncthreads();
            __builtin_amdgcn_s_cluster_barrier();  // wait layer1 finish step t
        } else {
            // ---- layer 1 phase A: recurrent half from h1[t-1] (overlaps L0) ----
            stage_tile(Ah, hprv_own, tid, wv);
            stage_wait(wv);
#pragma unroll 2
            for (int kt = 8; kt < 16; ++kt) {
                v16bf b = load_bfrag(Wl, (q*2 + nt)*16 + kt, lane);
#pragma unroll
                for (int m = 0; m < 4; ++m) {
                    v16bf a = load_afrag(Ah, m, kt & 7, lane);
                    acc[m] = __builtin_amdgcn_wmma_f32_16x16x32_bf16(
                        false, a, false, b, (short)0, acc[m], false, false);
                }
            }
            __syncthreads();
            cluster_barrier_acq();          // h0[t] now visible
            // ---- phase B: input half from h0[t] ----
            stage_tile(Ax, hcur0, tid, wv);
            stage_wait(wv);
#pragma unroll 2
            for (int kt = 0; kt < 8; ++kt) {
                v16bf b = load_bfrag(Wl, (q*2 + nt)*16 + kt, lane);
#pragma unroll
                for (int m = 0; m < 4; ++m) {
                    v16bf a = load_afrag(Ax, m, kt, lane);
                    acc[m] = __builtin_amdgcn_wmma_f32_16x16x32_bf16(
                        false, a, false, b, (short)0, acc[m], false, false);
                }
            }
            {
                int n = lane & 15, mh = lane >> 4;
#pragma unroll
                for (int m = 0; m < 4; ++m)
#pragma unroll
                    for (int r = 0; r < 8; ++r)
                        Gb[q*2048 + (m*16 + mh*8 + r)*32 + nt*16 + n] = acc[m][r];
            }
            __syncthreads();
            for (int i = tid; i < BATCH*32; i += 256) {
                int row = i >> 5, c = i & 31;
                int gcol = wgl*32 + c;
                float gi = Gb[0*2048 + i] + bl[0*256 + gcol];
                float gf = Gb[1*2048 + i] + bl[1*256 + gcol];
                float gg = Gb[2*2048 + i] + bl[2*256 + gcol];
                float go = Gb[3*2048 + i] + bl[3*256 + gcol];
                float cc = sigmoidf_(gf)*Cb[i] + sigmoidf_(gi)*tanhf(gg);
                Cb[i] = cc;
                __bf16 hb = (__bf16)(sigmoidf_(go)*tanhf(cc));
                hcur_own[row*HDIM + gcol] = hb;
                hs[(size_t)t*BATCH*HDIM + row*HDIM + gcol] = hb;
            }
            cluster_release();
            __syncthreads();
            __builtin_amdgcn_s_cluster_barrier();  // publish h1[t]
        }
    }
}

// ---------------------------------------------------------------------------
// Final: out = softmax(hs @ w_lin^T + b_lin).  2048 WGs x 64 rows.
// LDS: Wl 131072 + At 33792 + Y 67584 = 232448 B
// ---------------------------------------------------------------------------
__global__ __launch_bounds__(256, 1)
void linear_softmax_kernel(const __bf16* __restrict__ hs,
                           const __bf16* __restrict__ wlin,
                           const float* __restrict__ bias,
                           float* __restrict__ out) {
    extern __shared__ char smem[];
    __bf16* Wl = (__bf16*)smem;                     // 131072 B
    __bf16* At = (__bf16*)(smem + 131072);          // 33792 B
    float*  Y  = (float*)(smem + 131072 + 33792);   // 67584 B

    const int tid = threadIdx.x, lane = tid & 31, wv = tid >> 5;
    const float* blin = bias + 2048;

    {
        const u32x4* s = (const u32x4*)wlin;
        u32x4* d = (u32x4*)Wl;
        for (int i = tid; i < 8192; i += 256) d[i] = s[i];
    }
    stage_tile(At, hs + (size_t)blockIdx.x*64*256, tid, wv);
    stage_wait(wv);

    v8f acc[4][2] = {};
#pragma unroll 2
    for (int kt = 0; kt < 8; ++kt) {
        v16bf b0 = load_bfrag(Wl, (2*wv + 0)*8 + kt, lane);
        v16bf b1 = load_bfrag(Wl, (2*wv + 1)*8 + kt, lane);
#pragma unroll
        for (int m = 0; m < 4; ++m) {
            v16bf a = load_afrag(At, m, kt, lane);
            acc[m][0] = __builtin_amdgcn_wmma_f32_16x16x32_bf16(
                false, a, false, b0, (short)0, acc[m][0], false, false);
            acc[m][1] = __builtin_amdgcn_wmma_f32_16x16x32_bf16(
                false, a, false, b1, (short)0, acc[m][1], false, false);
        }
    }
    {
        int n = lane & 15, mh = lane >> 4;
#pragma unroll
        for (int m = 0; m < 4; ++m)
#pragma unroll
            for (int j = 0; j < 2; ++j)
#pragma unroll
                for (int r = 0; r < 8; ++r) {
                    int row = m*16 + mh*8 + r;
                    int col = (2*wv + j)*16 + n;
                    Y[row*264 + col] = acc[m][j][r] + blin[col];
                }
    }
    __syncthreads();
    // softmax: wave wv owns rows [wv*8, wv*8+8); lane handles 8 strided cols
    for (int rr = 0; rr < 8; ++rr) {
        int row = wv*8 + rr;
        float v[8];
        float mx = -__builtin_inff();
#pragma unroll
        for (int j = 0; j < 8; ++j) { v[j] = Y[row*264 + j*32 + lane]; mx = fmaxf(mx, v[j]); }
        for (int off = 16; off; off >>= 1) mx = fmaxf(mx, __shfl_xor(mx, off, 32));
        float sum = 0.f;
#pragma unroll
        for (int j = 0; j < 8; ++j) { v[j] = __expf(v[j] - mx); sum += v[j]; }
        for (int off = 16; off; off >>= 1) sum += __shfl_xor(sum, off, 32);
        float inv = 1.f / sum;
        float* orow = out + ((size_t)blockIdx.x*64 + row)*256;
#pragma unroll
        for (int j = 0; j < 8; ++j) orow[j*32 + lane] = v[j]*inv;
    }
}

// ---------------------------------------------------------------------------
extern "C" void kernel_launch(void* const* d_in, const int* in_sizes, int n_in,
                              void* d_out, int out_size, void* d_ws, size_t ws_size,
                              hipStream_t stream) {
    (void)in_sizes; (void)n_in; (void)out_size; (void)ws_size;
    const float* x     = (const float*)d_in[0];
    const float* w_ih0 = (const float*)d_in[1];
    const float* w_hh0 = (const float*)d_in[2];
    const float* b_ih0 = (const float*)d_in[3];
    const float* b_hh0 = (const float*)d_in[4];
    const float* w_ih1 = (const float*)d_in[5];
    const float* w_hh1 = (const float*)d_in[6];
    const float* b_ih1 = (const float*)d_in[7];
    const float* b_hh1 = (const float*)d_in[8];
    const float* w_lin = (const float*)d_in[9];
    const float* b_lin = (const float*)d_in[10];

    char* ws = (char*)d_ws;
    __bf16* xb   = (__bf16*)(ws + WS_XB);
    __bf16* hs   = (__bf16*)(ws + WS_HS);
    __bf16* wrec = (__bf16*)(ws + WS_WREC);
    __bf16* wlin = (__bf16*)(ws + WS_WLIN);
    float*  bias = (float*)(ws + WS_BIAS);
    __bf16* hbuf = (__bf16*)(ws + WS_HBUF);

    (void)hipFuncSetAttribute(reinterpret_cast<const void*>(&lstm_recurrent_kernel),
                              hipFuncAttributeMaxDynamicSharedMemorySize, 240*1024);
    (void)hipFuncSetAttribute(reinterpret_cast<const void*>(&linear_softmax_kernel),
                              hipFuncAttributeMaxDynamicSharedMemorySize, 236*1024);

    cvt_x_kernel<<<2048, 256, 0, stream>>>(x, xb, S_LEN*BATCH*IDIM);
    pack_rec_kernel<<<4096, 256, 0, stream>>>(w_ih0, w_hh0, w_ih1, w_hh1, wrec);
    pack_misc_kernel<<<256, 256, 0, stream>>>(w_lin, b_ih0, b_hh0, b_ih1, b_hh1, b_lin,
                                              wlin, bias);

    const size_t lds_rec = 131072 + 2*33792 + 32768 + 8192;   // 239616 B
    lstm_recurrent_kernel<<<16, 256, lds_rec, stream>>>(xb, hs, wrec, bias, hbuf);

    const size_t lds_fin = 131072 + 33792 + 67584;            // 232448 B
    linear_softmax_kernel<<<2048, 256, lds_fin, stream>>>(hs, wlin, bias, (float*)d_out);
}
